// MultiAspectRetrieval_77498389889522
// MI455X (gfx1250) — matrix operand order repaired and checked by reading, not compiled, not run on hardware
//
#include <hip/hip_runtime.h>

// ---------------------------------------------------------------------------
// MultiAspectRetrieval for MI455X (gfx1250, wave32, WMMA)
// B=256, N=8192, D=16384, D_A=512, S=4, D_K=64, C=S*D_K=256, K_MAX=64
// ---------------------------------------------------------------------------

typedef _Float16 v16h __attribute__((ext_vector_type(16)));
typedef _Float16 v8h  __attribute__((ext_vector_type(8)));
typedef float    v8f  __attribute__((ext_vector_type(8)));

constexpr int Bb = 256;     // batch
constexpr int Nn = 8192;    // pool size
constexpr int Dd = 16384;   // pool dim
constexpr int DA = 512;     // z dim
constexpr int Ss = 4;       // aspects
constexpr int DK = 64;      // key dim
constexpr int Cc = Ss * DK; // 256 fused columns

#define WMMA_F16(a, b, c) \
  __builtin_amdgcn_wmma_f32_16x16x32_f16(false, (a), false, (b), (short)0, (c), false, false)

__device__ __forceinline__ v16h mk16(const _Float16* p0, const _Float16* p1) {
  v8h lo = *(const v8h*)p0;
  v8h hi = *(const v8h*)p1;
  return __builtin_shufflevector(lo, hi, 0,1,2,3,4,5,6,7,8,9,10,11,12,13,14,15);
}

__device__ __forceinline__ unsigned f2s(float f) {
  unsigned u = __float_as_uint(f);
  return (u & 0x80000000u) ? ~u : (u | 0x80000000u);  // order-preserving map
}

// ---------------------------------------------------------------------------
// Kernel 1: queries = z @ W_Q[s], L2-normalize over D_K, scale by w[s].
// Output Qp [B, C] f16.  One block per batch element; 4 groups of 64 threads.
// ---------------------------------------------------------------------------
__global__ __launch_bounds__(256) void k_queries(const float* __restrict__ z,
                                                 const float* __restrict__ WQ,
                                                 const float* __restrict__ logits,
                                                 _Float16* __restrict__ Qp) {
  const int b = blockIdx.x, t = threadIdx.x;
  const int s = t >> 6, q = t & 63;
  const float* zrow = z + (size_t)b * DA;
  const float* w = WQ + ((size_t)s * DA) * DK + q;
  float acc = 0.f;
#pragma unroll 4
  for (int d = 0; d < DA; ++d) acc += zrow[d] * w[(size_t)d * DK];

  __shared__ float red[256];
  red[t] = acc * acc;
  __syncthreads();
#pragma unroll
  for (int off = 32; off > 0; off >>= 1) {
    if (q < off) red[t] += red[t + off];
    __syncthreads();
  }
  const float nrm = sqrtf(red[s << 6]);

  // softmax over 4 aspect logits
  const float l0 = logits[0], l1 = logits[1], l2 = logits[2], l3 = logits[3];
  const float m = fmaxf(fmaxf(l0, l1), fmaxf(l2, l3));
  const float e0 = expf(l0 - m), e1 = expf(l1 - m), e2 = expf(l2 - m), e3 = expf(l3 - m);
  const float lsel = (s == 0) ? l0 : (s == 1) ? l1 : (s == 2) ? l2 : l3;
  const float ws = expf(lsel - m) / (e0 + e1 + e2 + e3);

  Qp[(size_t)b * Cc + t] = (_Float16)(ws * acc / (nrm + 1e-8f));
}

// ---------------------------------------------------------------------------
// Kernel 2: keys GEMM  [8192 x 16384] x [16384 x 256] -> keys [N, C] fp32.
// f16 WMMA, f32 accumulate. 128 WGs x (64 rows, all 256 cols), KC=32/stage,
// double-buffered LDS (rows padded to 40 halves: conflict-free frag reads),
// register prefetch of next stage, batched fragment loads.
// ---------------------------------------------------------------------------
constexpr int MT = 64;   // rows per workgroup
constexpr int KC = 32;   // K per stage (== WMMA K)
constexpr int LP = 40;   // padded LDS row length in halves (80 B)

__global__ __launch_bounds__(256) void k_keys(const float* __restrict__ pool,
                                              const float* __restrict__ WK,
                                              float* __restrict__ keysOut) {
  __shared__ __align__(64) _Float16 As[2][MT][LP];   // 10 KB
  __shared__ __align__(64) _Float16 Bs[2][Cc][LP];   // 40 KB (col-major slice)

  const int t = threadIdx.x;
  const int n0 = blockIdx.x * MT;

  // A staging: thread -> (row ar, 8-float chunk aq)
  const int ar = t >> 2, aq = t & 3;
  // B staging: thread -> 8(k) x 4(c) sub-block: kg = k-group, cg = col-group
  const int kg = t & 3;        // k0 = kg*8
  const int cg = t >> 2;       // c0 = cg*4
  const int k0 = kg * 8;
  const int c0s = cg * 4;
  const int bs = c0s >> 6;     // aspect
  const int bq = c0s & 63;     // col within aspect

  const float* aBase = pool + (size_t)(n0 + ar) * Dd + aq * 8;
  const float* bBase = WK + ((size_t)bs * Dd + k0) * DK + bq;

  float areg[8];
  float breg[32];              // breg[dk*4 + j] : k = k0+dk, c = c0s+j
  auto loadStage = [&](int kk) {
    const float4* ap = (const float4*)(aBase + kk);
    float4 x0 = ap[0], x1 = ap[1];
    areg[0] = x0.x; areg[1] = x0.y; areg[2] = x0.z; areg[3] = x0.w;
    areg[4] = x1.x; areg[5] = x1.y; areg[6] = x1.z; areg[7] = x1.w;
#pragma unroll
    for (int dk = 0; dk < 8; ++dk) {
      float4 y = *(const float4*)(bBase + ((size_t)kk + dk) * DK);
      breg[4 * dk + 0] = y.x; breg[4 * dk + 1] = y.y;
      breg[4 * dk + 2] = y.z; breg[4 * dk + 3] = y.w;
    }
  };

  const int wave = t >> 5, lane = t & 31;
  const int m0 = (wave & 3) << 4;          // row sub-tile
  const int c0 = (wave >> 2) << 7;         // 128-col half
  const int lrow = lane & 15;
  const int koffA = (lane < 16) ? 0 : 8;   // 16-bit A-matrix lane layout
  const int koffB = (lane < 16) ? 0 : 8;   // 16-bit B-matrix lane layout

  v8f acc[8] = {};
  int buf = 0;
  loadStage(0);
  const int NST = Dd / KC;  // 512 stages
  for (int st = 0; st < NST; ++st) {
    // convert staged regs -> LDS (f32 -> f16)
    v8h av;
#pragma unroll
    for (int e = 0; e < 8; ++e) av[e] = (_Float16)areg[e];
    *(v8h*)&As[buf][ar][aq * 8] = av;
#pragma unroll
    for (int j = 0; j < 4; ++j) {       // one b128 store per output column
      v8h bv;
#pragma unroll
      for (int dk = 0; dk < 8; ++dk) bv[dk] = (_Float16)breg[4 * dk + j];
      *(v8h*)&Bs[buf][c0s + j][k0] = bv;
    }
    __syncthreads();

    if (st + 1 < NST) loadStage((st + 1) * KC);           // overlap next loads
    if (st + 2 < NST) __builtin_prefetch(aBase + (size_t)(st + 2) * KC, 0, 0);

    // batch all fragment loads into distinct registers, then WMMA chain
    const v16h a = mk16(&As[buf][m0 + lrow][koffA], &As[buf][m0 + lrow][16 + koffA]);
    v16h bfr[8];
#pragma unroll
    for (int j = 0; j < 8; ++j)
      bfr[j] = mk16(&Bs[buf][c0 + (j << 4) + lrow][koffB],
                    &Bs[buf][c0 + (j << 4) + lrow][koffB + 16]);
#pragma unroll
    for (int j = 0; j < 8; ++j) acc[j] = WMMA_F16(a, bfr[j], acc[j]);
    __syncthreads();
    buf ^= 1;
  }

  // C/D layout: VGPR r -> row m0+r (lanes 0-15) / m0+8+r (lanes 16-31), col = lane%16
  const int rowBase = n0 + m0 + ((lane < 16) ? 0 : 8);
#pragma unroll
  for (int j = 0; j < 8; ++j) {
    const int c = c0 + (j << 4) + lrow;
#pragma unroll
    for (int r = 0; r < 8; ++r)
      keysOut[(size_t)(rowBase + r) * Cc + c] = acc[j][r];
  }
}

// ---------------------------------------------------------------------------
// Kernel 3: normalize keys per (n,s) over D_K -> Kp [N, C] f16.
// ---------------------------------------------------------------------------
__global__ __launch_bounds__(256) void k_knorm(const float* __restrict__ keys,
                                               _Float16* __restrict__ Kp) {
  const int n = blockIdx.x, t = threadIdx.x;
  const int s = t >> 6, q = t & 63;
  const float v = keys[(size_t)n * Cc + t];
  __shared__ float red[256];
  red[t] = v * v;
  __syncthreads();
#pragma unroll
  for (int off = 32; off > 0; off >>= 1) {
    if (q < off) red[t] += red[t + off];
    __syncthreads();
  }
  const float nrm = sqrtf(red[s << 6]);
  Kp[(size_t)n * Cc + t] = (_Float16)(v / (nrm + 1e-8f));
}

// ---------------------------------------------------------------------------
// Kernel 4: scores = Qp @ Kp^T  ([256 x 256] x [256 x 8192]) via WMMA.
// Grid (N/64, B/64); A-fragments live in registers, B tile in LDS with rows
// padded to 264 halves (33 uint4) for conflict-free fragment reads.
// ---------------------------------------------------------------------------
constexpr int KP4 = Cc / 8;       // 32 uint4 per source row
constexpr int BKP = KP4 + 1;      // 33 uint4 per padded LDS row

__global__ __launch_bounds__(256) void k_scores(const _Float16* __restrict__ Qp,
                                                const _Float16* __restrict__ Kp,
                                                float* __restrict__ scores) {
  __shared__ __align__(64) _Float16 Bk[64 * BKP * 8];  // 33 KB: row c = Kp row n0+c
  const int t = threadIdx.x;
  const int n0 = blockIdx.x * 64;   // pool cols
  const int r0 = blockIdx.y * 64;   // batch rows

  {  // copy 64 Kp rows into padded LDS rows
    const uint4* src = (const uint4*)(Kp + (size_t)n0 * Cc);
    uint4* dst = (uint4*)Bk;
#pragma unroll
    for (int i = 0; i < 8; ++i) {
      const int idx = t + 256 * i;             // 0..2047
      const int r = idx >> 5, c16 = idx & 31;  // src row, 16B chunk
      dst[r * BKP + c16] = src[r * KP4 + c16];
    }
  }

  const int wave = t >> 5, lane = t & 31;
  const int rw = r0 + ((wave & 3) << 4);   // 16-row sub-tile
  const int cw = (wave >> 2) << 5;         // 32-col half
  const int lrow = lane & 15;
  const int koffA = (lane < 16) ? 0 : 8;
  const int koffB = (lane < 16) ? 0 : 8;

  // Preload all 8 A fragments (K=256) from global (L2-resident) into registers
  v16h a[8];
  const _Float16* arow = Qp + (size_t)(rw + lrow) * Cc;
#pragma unroll
  for (int ks = 0; ks < 8; ++ks)
    a[ks] = mk16(arow + ks * 32 + koffA, arow + ks * 32 + 16 + koffA);

  __syncthreads();

  v8f acc[2] = {};
#pragma unroll
  for (int ks = 0; ks < 8; ++ks) {
    v16h bfr[2];
#pragma unroll
    for (int j = 0; j < 2; ++j) {
      const int c = cw + (j << 4) + lrow;
      const _Float16* bp = Bk + (size_t)c * (BKP * 8) + ks * 32 + koffB;
      bfr[j] = mk16(bp, bp + 16);
    }
#pragma unroll
    for (int j = 0; j < 2; ++j) acc[j] = WMMA_F16(a[ks], bfr[j], acc[j]);
  }

  const int bRow = rw + ((lane < 16) ? 0 : 8);
#pragma unroll
  for (int j = 0; j < 2; ++j) {
    const int n = n0 + cw + (j << 4) + lrow;
#pragma unroll
    for (int r = 0; r < 8; ++r)
      scores[(size_t)(bRow + r) * Nn + n] = acc[j][r];
  }
}

// ---------------------------------------------------------------------------
// Kernel 5: per-row top-k threshold (radix binary search), masked scores,
// sigmoid-gated exp weights, normalized alpha. One block per batch row.
// ---------------------------------------------------------------------------
__global__ __launch_bounds__(256) void k_topk_alpha(const float* __restrict__ tauP,
                                                    const float* __restrict__ lamP,
                                                    const float* __restrict__ tempP,
                                                    const int* __restrict__ kmaxP,
                                                    float* __restrict__ scoresSeg,
                                                    float* __restrict__ alphaSeg) {
  const int b = blockIdx.x, t = threadIdx.x;
  __shared__ float sv[Nn];  // 32 KB
  __shared__ unsigned cntS;
  __shared__ float ssum;

  const float tau = *tauP, lam = *lamP, temp = *tempP;
  const unsigned kmax = (unsigned)*kmaxP;

  float* row = scoresSeg + (size_t)b * Nn;
  for (int i = t; i < Nn; i += 256) sv[i] = row[i];
  __syncthreads();

  // Largest threshold T (in sortable-uint space) with count(u >= T) >= kmax
  unsigned T = 0;
  if ((int)kmax < Nn) {
    for (int bit = 31; bit >= 0; --bit) {
      const unsigned cand = T | (1u << bit);
      if (t == 0) cntS = 0;
      __syncthreads();
      unsigned c = 0;
      for (int i = t; i < Nn; i += 256) c += (f2s(sv[i]) >= cand) ? 1u : 0u;
      atomicAdd(&cntS, c);
      __syncthreads();
      const unsigned tot = cntS;
      __syncthreads();
      if (tot >= kmax) T = cand;
    }
  }

  if (t == 0) ssum = 0.f;
  __syncthreads();
  float psum = 0.f;
  for (int i = t; i < Nn; i += 256) {
    const float s = sv[i];
    const bool keep = ((int)kmax >= Nn) || (f2s(s) >= T);
    const float sm = keep ? s : -1e9f;
    const float g = 1.f / (1.f + expf(-lam * (sm - tau)));  // -> 0 for masked
    const float ar = g * expf(sm / temp);                   // -> 0 for masked
    row[i] = sm;      // masked scores output
    sv[i] = ar;
    psum += ar;
  }
  atomicAdd(&ssum, psum);
  __syncthreads();
  const float inv = 1.f / (ssum + 1e-8f);
  for (int i = t; i < Nn; i += 256) alphaSeg[(size_t)b * Nn + i] = sv[i] * inv;
}

// ---------------------------------------------------------------------------
// Host launcher
// ---------------------------------------------------------------------------
extern "C" void kernel_launch(void* const* d_in, const int* in_sizes, int n_in,
                              void* d_out, int out_size, void* d_ws, size_t ws_size,
                              hipStream_t stream) {
  (void)in_sizes; (void)n_in; (void)out_size; (void)ws_size;
  const float* z      = (const float*)d_in[0];
  const float* pool   = (const float*)d_in[1];
  const float* WQ     = (const float*)d_in[2];
  const float* WK     = (const float*)d_in[3];
  const float* logits = (const float*)d_in[4];
  const float* tau    = (const float*)d_in[5];
  const float* lam    = (const float*)d_in[6];
  const float* temp   = (const float*)d_in[7];
  const int*   kmax   = (const int*)d_in[8];

  float* alphaSeg  = (float*)d_out;                         // [B,N]
  float* scoresSeg = alphaSeg + (size_t)Bb * Nn;            // [B,N]
  float* keysSeg   = scoresSeg + (size_t)Bb * Nn;           // [N,S,DK] == [N,C]

  _Float16* Qp = (_Float16*)d_ws;                           // [B,C]  128 KB
  _Float16* Kp = Qp + (size_t)Bb * Cc;                      // [N,C]  4 MB

  k_queries<<<Bb, 256, 0, stream>>>(z, WQ, logits, Qp);
  k_keys<<<Nn / MT, 256, 0, stream>>>(pool, WK, keysSeg);
  k_knorm<<<Nn, 256, 0, stream>>>(keysSeg, Kp);
  k_scores<<<dim3(Nn / 64, Bb / 64), 256, 0, stream>>>(Qp, Kp, scoresSeg);
  k_topk_alpha<<<Bb, 256, 0, stream>>>(tau, lam, temp, kmax, scoresSeg, alphaSeg);
}
